// TreeLSTMCellDP_73830487818705
// MI455X (gfx1250) — compile-verified, matrix operand order checked by reading, not run on hardware
//
#include <hip/hip_runtime.h>
#include <hip/hip_bf16.h>
#include <stdint.h>

// ---------------- problem constants ----------------
#define NROWS 4096
#define HDIM  256
#define TWOH  512
#define THREEH 768
#define ND1 3
#define ND2 2
#define ND3 2
#define CELLS 12            // ND1*ND2*ND3
#define SLICES 24           // 2 sides * CELLS
#define MTOT  49152         // CELLS * NROWS
#define BOTF  1e30f

// ---------------- WMMA types ----------------
typedef __attribute__((ext_vector_type(16))) __bf16 v16bf;
typedef __attribute__((ext_vector_type(8)))  float  v8f;

// ---------------- helpers ----------------
__device__ __forceinline__ unsigned short f32_to_bf16(float f) {
    return __builtin_bit_cast(unsigned short, (__bf16)f);     // native v_cvt on gfx1250
}
__device__ __forceinline__ unsigned int pack_bf16x2(float a, float b) {
    unsigned int lo = f32_to_bf16(a);
    unsigned int hi = f32_to_bf16(b);
    return (hi << 16) | lo;
}
__device__ __forceinline__ float bf16_to_f32(unsigned short s) {
    return __uint_as_float(((unsigned int)s) << 16);
}
__device__ __forceinline__ float sigmoidf(float x) { return 1.0f / (1.0f + __expf(-x)); }

struct IV { float v, l, u; };
__device__ __forceinline__ IV imul(IV a, IV b) {
    float p1 = a.l * b.l, p2 = a.l * b.u, p3 = a.u * b.l, p4 = a.u * b.u;
    IV r;
    r.v = a.v * b.v;
    r.l = fminf(fminf(p1, p2), fminf(p3, p4));
    r.u = fmaxf(fmaxf(p1, p2), fmaxf(p3, p4));
    return r;
}
__device__ __forceinline__ IV iadd(IV a, IV b) { return {a.v + b.v, a.l + b.l, a.u + b.u}; }
__device__ __forceinline__ IV imerge(IV a, IV b) {
    bool bot = a.l > a.u;
    IV r;
    r.v = bot ? b.v : a.v;
    r.l = fminf(a.l, b.l);
    r.u = fmaxf(a.u, b.u);
    return r;
}

// ---------------- kernel 0: weights f32 -> bf16 ----------------
__global__ void k_convert_weights(const float* __restrict__ UfW,
                                  const float* __restrict__ UiouW,
                                  unsigned short* __restrict__ wf,
                                  unsigned short* __restrict__ wiou) {
    int i = blockIdx.x * blockDim.x + threadIdx.x;
    if (i < TWOH * TWOH)   wf[i]   = f32_to_bf16(UfW[i]);
    if (i < THREEH * TWOH) wiou[i] = f32_to_bf16(UiouW[i]);
}

// ---------------- GEMM tiling ----------------
// Block: 256 threads = 8 waves arranged 4(M) x 2(N).
// Each wave: 16M x 32N (two 16x16 N-subtiles) x 3 interval components
//   -> 6 wmma per K-step from 7 fragment loads (A reused across N-subtiles,
//      B tiles reused across components).
#define BM 64
#define BN 64
#define BK 32

// kernel 1: per-slice partial products P{v,c,r}[slice] = X_slice{v,cen,rad} @ W_half{W,|W|}.T
// P layout: [comp][slice][row][col] bf16, comp stride PC
__global__ void __launch_bounds__(256) k_gemm_f(
    const float* __restrict__ hv, const float* __restrict__ hlb, const float* __restrict__ hub,
    const unsigned short* __restrict__ wf,
    unsigned short* __restrict__ P)
{
    __shared__ unsigned short As[3][BM][BK + 2];
    __shared__ unsigned short Bs[2][BN][BK + 2];

    const int tid = threadIdx.x;
    const int rowBase = blockIdx.x * BM;
    const int nBase   = blockIdx.y * BN;
    const int slice   = blockIdx.z;
    const int side = slice / CELLS;
    const int ci   = slice % CELLS;

    const int lane = tid & 31;
    const int wave = tid >> 5;
    const int wm  = (wave >> 1) * 16;   // 0,16,32,48
    const int wnw = (wave & 1) * 32;    // 0,32
    const int mm = lane & 15;
    const int hs = lane >> 4;

    v8f acc[3][2];
    #pragma unroll
    for (int c = 0; c < 3; ++c)
        #pragma unroll
        for (int s = 0; s < 2; ++s) acc[c][s] = (v8f){};

    for (int k0 = 0; k0 < HDIM; k0 += BK) {
        // ---- stage A (v,cen,rad) : 64 rows x 32 k, float2 loads, packed bf16x2 stores
        for (int p = tid; p < BM * (BK / 2); p += 256) {
            int r  = p >> 4;
            int kp = (p & 15) * 2;
            size_t g = (((size_t)(rowBase + r) * 2 + side) * CELLS + ci) * HDIM + (k0 + kp);
            float2 v  = *(const float2*)(hv  + g);
            float2 lb = *(const float2*)(hlb + g);
            float2 ub = *(const float2*)(hub + g);
            *(unsigned int*)&As[0][r][kp] = pack_bf16x2(v.x, v.y);
            *(unsigned int*)&As[1][r][kp] = pack_bf16x2(0.5f * (lb.x + ub.x), 0.5f * (lb.y + ub.y));
            *(unsigned int*)&As[2][r][kp] = pack_bf16x2(0.5f * (ub.x - lb.x), 0.5f * (ub.y - lb.y));
        }
        // ---- stage B (W, |W|) : 64 cols x 32 k, u32 loads/stores
        for (int p = tid; p < BN * (BK / 2); p += 256) {
            int n  = p >> 4;
            int kp = (p & 15) * 2;
            unsigned int w2 = *(const unsigned int*)(wf + (size_t)(nBase + n) * TWOH + side * HDIM + k0 + kp);
            *(unsigned int*)&Bs[0][n][kp] = w2;
            *(unsigned int*)&Bs[1][n][kp] = w2 & 0x7FFF7FFFu;
        }
        __syncthreads();

        union FR { v16bf v; unsigned short u[16]; };
        FR a[3], b[2][2];
        #pragma unroll
        for (int q = 0; q < 8; ++q) {
            int kb = ((q >> 2) * 16) + hs * 8 + ((q & 3) * 2);
            #pragma unroll
            for (int c = 0; c < 3; ++c) {
                a[c].u[2*q]   = As[c][wm + mm][kb];
                a[c].u[2*q+1] = As[c][wm + mm][kb + 1];
            }
            #pragma unroll
            for (int vvar = 0; vvar < 2; ++vvar)
                #pragma unroll
                for (int s = 0; s < 2; ++s) {
                    b[vvar][s].u[2*q]   = Bs[vvar][wnw + s * 16 + mm][kb];
                    b[vvar][s].u[2*q+1] = Bs[vvar][wnw + s * 16 + mm][kb + 1];
                }
        }
        #pragma unroll
        for (int s = 0; s < 2; ++s) {
            acc[0][s] = __builtin_amdgcn_wmma_f32_16x16x32_bf16(false, a[0].v, false, b[0][s].v, (short)0, acc[0][s], false, false);
            acc[1][s] = __builtin_amdgcn_wmma_f32_16x16x32_bf16(false, a[1].v, false, b[0][s].v, (short)0, acc[1][s], false, false);
            acc[2][s] = __builtin_amdgcn_wmma_f32_16x16x32_bf16(false, a[2].v, false, b[1][s].v, (short)0, acc[2][s], false, false);
        }
        __syncthreads();
    }

    const size_t PC = (size_t)SLICES * NROWS * TWOH;
    #pragma unroll
    for (int s = 0; s < 2; ++s)
        #pragma unroll
        for (int e = 0; e < 8; ++e) {
            int mo = e + hs * 8;
            size_t o = ((size_t)slice * NROWS + (rowBase + wm + mo)) * TWOH + (nBase + wnw + s * 16 + mm);
            P[o]          = f32_to_bf16(acc[0][s][e]);
            P[PC + o]     = f32_to_bf16(acc[1][s][e]);
            P[2 * PC + o] = f32_to_bf16(acc[2][s][e]);
        }
}

// ---------------- kernel 2: fold combos, build new_h_cat / new_c ----------------
__global__ void __launch_bounds__(256) k_combine(
    const float* __restrict__ hv, const float* __restrict__ hlb, const float* __restrict__ hub,
    const float* __restrict__ cv, const float* __restrict__ clb, const float* __restrict__ cub,
    const float* __restrict__ Ufb,
    const unsigned short* __restrict__ P,
    unsigned short* __restrict__ newh,   // [comp:{v,cen,rad}][MTOT][TWOH] bf16
    float* __restrict__ newc)            // [comp:{v,l,u}][MTOT][HDIM] f32
{
    const int row = blockIdx.x;
    const int ci  = blockIdx.y;
    const int hh  = threadIdx.x;
    const int d1 = ci >> 2, d2 = (ci >> 1) & 1, d3 = ci & 1;

    IV ph0 = {0.f, BOTF, -BOTF};
    IV ph1 = ph0;
    IV pc  = ph0;

    const size_t PC = (size_t)SLICES * NROWS * TWOH;

    for (int l1 = 0; l1 <= d1; ++l1)
    for (int l2 = 0; l2 <= d2; ++l2)
    for (int l3 = 0; l3 <= d3; ++l3) {
        int cil = (l1 << 2) | (l2 << 1) | l3;
        int cir = ((d1 - l1) << 2) | ((d2 - l2) << 1) | (d3 - l3);
        int sl = cil;            // side-0 slice
        int sr = CELLS + cir;    // side-1 slice

        IV f[2];
        #pragma unroll
        for (int t = 0; t < 2; ++t) {
            int j = hh + t * HDIM;
            size_t ol  = ((size_t)sl * NROWS + row) * TWOH + j;
            size_t orr = ((size_t)sr * NROWS + row) * TWOH + j;
            float sv  = bf16_to_f32(P[ol]) + bf16_to_f32(P[orr]) + Ufb[j];
            float sc  = bf16_to_f32(P[PC + ol]) + bf16_to_f32(P[PC + orr]) + Ufb[j];
            float sr_ = bf16_to_f32(P[2 * PC + ol]) + bf16_to_f32(P[2 * PC + orr]);
            f[t].v = sigmoidf(sv);
            f[t].l = sigmoidf(sc - sr_);
            f[t].u = sigmoidf(sc + sr_);
        }

        size_t gl = ((size_t)(row * 2 + 0) * CELLS + cil) * HDIM + hh;
        size_t gr = ((size_t)(row * 2 + 1) * CELLS + cir) * HDIM + hh;
        IV cl = { cv[gl], clb[gl], cub[gl] };
        IV cr = { cv[gr], clb[gr], cub[gr] };
        IV cc = iadd(imul(f[0], cl), imul(f[1], cr));
        pc = imerge(pc, cc);

        IV hl = { hv[gl], hlb[gl], hub[gl] };
        IV hr = { hv[gr], hlb[gr], hub[gr] };
        ph0 = imerge(ph0, hl);
        ph1 = imerge(ph1, hr);
    }

    const size_t NH = (size_t)MTOT * TWOH;
    size_t ob = ((size_t)(ci * NROWS + row)) * TWOH;
    newh[ob + hh]                 = f32_to_bf16(ph0.v);
    newh[NH + ob + hh]            = f32_to_bf16(0.5f * (ph0.l + ph0.u));
    newh[2 * NH + ob + hh]        = f32_to_bf16(0.5f * (ph0.u - ph0.l));
    newh[ob + HDIM + hh]          = f32_to_bf16(ph1.v);
    newh[NH + ob + HDIM + hh]     = f32_to_bf16(0.5f * (ph1.l + ph1.u));
    newh[2 * NH + ob + HDIM + hh] = f32_to_bf16(0.5f * (ph1.u - ph1.l));

    const size_t NC = (size_t)MTOT * HDIM;
    size_t oc = ((size_t)(ci * NROWS + row)) * HDIM + hh;
    newc[oc]          = pc.v;
    newc[NC + oc]     = pc.l;
    newc[2 * NC + oc] = pc.u;
}

// ---------------- kernel 3: iou GEMM with fused sigmoid interval epilogue ----------------
// A: newh [3][MTOT][512] bf16, B: Uiou (768x512) bf16, out: sigmoid(iou) [3][MTOT][768] bf16
__global__ void __launch_bounds__(256) k_gemm_iou(
    const unsigned short* __restrict__ A,
    const unsigned short* __restrict__ wiou,
    const float* __restrict__ biou,
    unsigned short* __restrict__ iou)
{
    __shared__ unsigned short As[3][BM][BK + 2];
    __shared__ unsigned short Bs[2][BN][BK + 2];

    const int tid = threadIdx.x;
    const int rowBase = blockIdx.x * BM;
    const int nBase   = blockIdx.y * BN;

    const int lane = tid & 31;
    const int wave = tid >> 5;
    const int wm  = (wave >> 1) * 16;
    const int wnw = (wave & 1) * 32;
    const int mm = lane & 15;
    const int hs = lane >> 4;

    v8f acc[3][2];
    #pragma unroll
    for (int c = 0; c < 3; ++c)
        #pragma unroll
        for (int s = 0; s < 2; ++s) acc[c][s] = (v8f){};

    const size_t AC = (size_t)MTOT * TWOH;

    for (int k0 = 0; k0 < TWOH; k0 += BK) {
        for (int p = tid; p < BM * (BK / 2); p += 256) {
            int r  = p >> 4;
            int kp = (p & 15) * 2;
            size_t g = (size_t)(rowBase + r) * TWOH + k0 + kp;
            *(unsigned int*)&As[0][r][kp] = *(const unsigned int*)(A + g);
            *(unsigned int*)&As[1][r][kp] = *(const unsigned int*)(A + AC + g);
            *(unsigned int*)&As[2][r][kp] = *(const unsigned int*)(A + 2 * AC + g);
        }
        for (int p = tid; p < BN * (BK / 2); p += 256) {
            int n  = p >> 4;
            int kp = (p & 15) * 2;
            unsigned int w2 = *(const unsigned int*)(wiou + (size_t)(nBase + n) * TWOH + k0 + kp);
            *(unsigned int*)&Bs[0][n][kp] = w2;
            *(unsigned int*)&Bs[1][n][kp] = w2 & 0x7FFF7FFFu;
        }
        __syncthreads();

        union FR { v16bf v; unsigned short u[16]; };
        FR a[3], b[2][2];
        #pragma unroll
        for (int q = 0; q < 8; ++q) {
            int kb = ((q >> 2) * 16) + hs * 8 + ((q & 3) * 2);
            #pragma unroll
            for (int c = 0; c < 3; ++c) {
                a[c].u[2*q]   = As[c][wm + mm][kb];
                a[c].u[2*q+1] = As[c][wm + mm][kb + 1];
            }
            #pragma unroll
            for (int vvar = 0; vvar < 2; ++vvar)
                #pragma unroll
                for (int s = 0; s < 2; ++s) {
                    b[vvar][s].u[2*q]   = Bs[vvar][wnw + s * 16 + mm][kb];
                    b[vvar][s].u[2*q+1] = Bs[vvar][wnw + s * 16 + mm][kb + 1];
                }
        }
        #pragma unroll
        for (int s = 0; s < 2; ++s) {
            acc[0][s] = __builtin_amdgcn_wmma_f32_16x16x32_bf16(false, a[0].v, false, b[0][s].v, (short)0, acc[0][s], false, false);
            acc[1][s] = __builtin_amdgcn_wmma_f32_16x16x32_bf16(false, a[1].v, false, b[0][s].v, (short)0, acc[1][s], false, false);
            acc[2][s] = __builtin_amdgcn_wmma_f32_16x16x32_bf16(false, a[2].v, false, b[1][s].v, (short)0, acc[2][s], false, false);
        }
        __syncthreads();
    }

    const size_t IO = (size_t)MTOT * THREEH;
    #pragma unroll
    for (int s = 0; s < 2; ++s)
        #pragma unroll
        for (int e = 0; e < 8; ++e) {
            int mo  = e + hs * 8;
            int col = nBase + wnw + s * 16 + mm;
            float bb = biou[col];
            size_t o = (size_t)(rowBase + wm + mo) * THREEH + col;
            iou[o]          = f32_to_bf16(sigmoidf(acc[0][s][e] + bb));
            iou[IO + o]     = f32_to_bf16(sigmoidf(acc[1][s][e] - acc[2][s][e] + bb));
            iou[2 * IO + o] = f32_to_bf16(sigmoidf(acc[1][s][e] + acc[2][s][e] + bb));
        }
}

// ---------------- kernel 4: final combine + aux merges + output ----------------
__global__ void __launch_bounds__(256) k_final(
    const float* __restrict__ hv, const float* __restrict__ hlb, const float* __restrict__ hub,
    const float* __restrict__ cv, const float* __restrict__ clb, const float* __restrict__ cub,
    const int* __restrict__ unk,
    const unsigned short* __restrict__ iou,
    const float* __restrict__ newc,
    float* __restrict__ out)
{
    const int row = blockIdx.x, ci = blockIdx.y, hh = threadIdx.x;
    const int d1 = ci >> 2, d2 = (ci >> 1) & 1, d3 = ci & 1;

    const size_t IO = (size_t)MTOT * THREEH;
    size_t mb = ((size_t)(ci * NROWS + row)) * THREEH;
    IV iv = { bf16_to_f32(iou[mb + hh]),
              bf16_to_f32(iou[IO + mb + hh]),
              bf16_to_f32(iou[2 * IO + mb + hh]) };
    IV ov = { bf16_to_f32(iou[mb + HDIM + hh]),
              bf16_to_f32(iou[IO + mb + HDIM + hh]),
              bf16_to_f32(iou[2 * IO + mb + HDIM + hh]) };
    IV uv = { bf16_to_f32(iou[mb + 2 * HDIM + hh]),
              bf16_to_f32(iou[IO + mb + 2 * HDIM + hh]),
              bf16_to_f32(iou[2 * IO + mb + 2 * HDIM + hh]) };

    const size_t NC = (size_t)MTOT * HDIM;
    size_t oc = ((size_t)(ci * NROWS + row)) * HDIM + hh;
    IV nc = { newc[oc], newc[NC + oc], newc[2 * NC + oc] };

    IV co = iadd(imul(iv, uv), nc);
    IV th = { tanhf(co.v), tanhf(co.l), tanhf(co.u) };
    IV ho = imul(ov, th);

    IV av = {0.f, BOTF, -BOTF};
    IV ac = av;
    #pragma unroll
    for (int j = 0; j < 2; ++j) {
        int um = unk[row * 2 + j];
        bool m = (um > 0) && (um <= d1);
        int id = d1 - um;
        id = id < 0 ? 0 : (id > (ND1 - 1) ? (ND1 - 1) : id);
        size_t g = ((size_t)(row * 2 + (1 - j)) * CELLS + ((id << 2) | (d2 << 1) | d3)) * HDIM + hh;
        IV oh  = { hv[g], hlb[g], hub[g] };
        IV occ = { cv[g], clb[g], cub[g] };
        bool both = av.l > av.u;
        if (m && both) av.v = oh.v;
        if (m) { av.l = fminf(av.l, oh.l); av.u = fmaxf(av.u, oh.u); }
        bool botc = ac.l > ac.u;
        if (m && botc) ac.v = occ.v;
        if (m) { ac.l = fminf(ac.l, occ.l); ac.u = fmaxf(ac.u, occ.u); }
    }

    IV hf = imerge(ho, av);
    IV cf = imerge(co, ac);

    const size_t OS = (size_t)NROWS * CELLS * HDIM;
    size_t oo = (((size_t)row) * CELLS + ci) * HDIM + hh;
    out[oo]          = hf.v;
    out[OS + oo]     = hf.l;
    out[2 * OS + oo] = hf.u;
    out[3 * OS + oo] = cf.v;
    out[4 * OS + oo] = cf.l;
    out[5 * OS + oo] = cf.u;
}

// ---------------- kernel 5: new_unk (int32 tail of output) ----------------
__global__ void k_unk(const int* __restrict__ unk, float* __restrict__ out) {
    int i = blockIdx.x * blockDim.x + threadIdx.x;
    if (i >= NROWS) return;
    int u0 = unk[2 * i], u1 = unk[2 * i + 1];
    int* o = (int*)(out + (size_t)6 * NROWS * CELLS * HDIM);
    o[i] = (u0 > 0 && u1 > 0) ? (u0 + u1) : 0;
}

// ---------------- launcher ----------------
extern "C" void kernel_launch(void* const* d_in, const int* in_sizes, int n_in,
                              void* d_out, int out_size, void* d_ws, size_t ws_size,
                              hipStream_t stream) {
    const float* hv   = (const float*)d_in[0];
    const float* hlb  = (const float*)d_in[1];
    const float* hub  = (const float*)d_in[2];
    const float* cv   = (const float*)d_in[3];
    const float* clb  = (const float*)d_in[4];
    const float* cub  = (const float*)d_in[5];
    const int*   unk  = (const int*)d_in[6];
    const float* UfW  = (const float*)d_in[7];
    const float* Ufb  = (const float*)d_in[8];
    const float* UiouW= (const float*)d_in[9];
    const float* biou = (const float*)d_in[10];
    float* out = (float*)d_out;

    // workspace layout (bytes); P region is dead after k_combine, so it aliases the iou region
    char* ws = (char*)d_ws;
    const size_t SZ_WF     = (size_t)TWOH * TWOH * 2;            // 512 KB
    const size_t SZ_WIOU   = (size_t)THREEH * TWOH * 2;          // 768 KB
    const size_t SZ_SHARED = (size_t)3 * MTOT * THREEH * 2;      // 216 MB (< P's 288 MB)
    const size_t SZ_NEWH   = (size_t)3 * MTOT * TWOH * 2;        // 144 MB
    // newc: 3 * MTOT * HDIM * 4 = 144 MB; total ~506 MB of d_ws
    unsigned short* wf    = (unsigned short*)(ws);
    unsigned short* wiou  = (unsigned short*)(ws + SZ_WF);
    unsigned short* P     = (unsigned short*)(ws + SZ_WF + SZ_WIOU);           // aliased
    unsigned short* iou   = (unsigned short*)(ws + SZ_WF + SZ_WIOU);           // aliased
    unsigned short* newh  = (unsigned short*)(ws + SZ_WF + SZ_WIOU + SZ_SHARED);
    float*          newc  = (float*)(ws + SZ_WF + SZ_WIOU + SZ_SHARED + SZ_NEWH);

    // 0) weights -> bf16
    k_convert_weights<<<dim3((THREEH * TWOH + 255) / 256), 256, 0, stream>>>(UfW, UiouW, wf, wiou);

    // 1) 24-slice partial GEMMs (M=4096, K=256, N=512), 3 interval comps each
    k_gemm_f<<<dim3(NROWS / BM, TWOH / BN, SLICES), 256, 0, stream>>>(hv, hlb, hub, wf, P);

    // 2) fold combos -> new_h_cat (bf16 v/cen/rad) and new_c
    k_combine<<<dim3(NROWS, CELLS), 256, 0, stream>>>(hv, hlb, hub, cv, clb, cub, Ufb, P, newh, newc);

    // 3) iou GEMM (M=49152, K=512, N=768) with fused bias+sigmoid interval epilogue
    k_gemm_iou<<<dim3(MTOT / BM, THREEH / BN), 256, 0, stream>>>(newh, wiou, biou, iou);

    // 4) final combine + aux merges
    k_final<<<dim3(NROWS, CELLS), 256, 0, stream>>>(hv, hlb, hub, cv, clb, cub, unk, iou, newc, out);

    // 5) new_unk tail
    k_unk<<<dim3((NROWS + 255) / 256), 256, 0, stream>>>(unk, out);
}
